// BT4_78743930405307
// MI455X (gfx1250) — compile-verified, matrix operand order checked by reading, not run on hardware
//
#include <hip/hip_runtime.h>
#include <hip/hip_bf16.h>

// ---------------------------------------------------------------------------
// CDNA5 (gfx1250) WMMA implementation of the 15-layer smolgen encoder stack.
// All heavy GEMMs run through v_wmma_f32_16x16x32_f16 with f32 accumulation.
// 128x128 block tile / 8 waves, each wave = 64x32 (8 WMMAs per 32-deep kstep).
// ---------------------------------------------------------------------------

typedef __attribute__((ext_vector_type(16))) _Float16 v16h;
typedef __attribute__((ext_vector_type(8)))  _Float16 v8h;
typedef __attribute__((ext_vector_type(4)))  _Float16 v4h;
typedef __attribute__((ext_vector_type(8)))  float    v8f;
typedef __attribute__((ext_vector_type(4)))  int      v4i;

#define L_LAYERS 15
#define E_DIM    1024
#define D_DIM    1024
#define H_HEADS  32
#define HD_DIM   32
#define DFF_DIM  1536
#define SC_DIM   32
#define SH_DIM   256
#define SG_DIM   256
#define SEQ_LEN  64
#define B_BATCH  128
#define ROWS     (B_BATCH * SEQ_LEN)          /* 8192 token rows            */
#define ALPHA_RES 0.4272870319f               /* (2*15)^-0.25               */
#define QK_SCALE  0.1767766953f               /* 1/sqrt(32)                 */

// Async global->LDS path (gfx1250 GLOBAL_LOAD_ASYNC_TO_LDS_B128, ASYNCcnt).
#if defined(__gfx1250__) &&                                            \
    __has_builtin(__builtin_amdgcn_global_load_async_to_lds_b128) &&   \
    __has_builtin(__builtin_amdgcn_s_wait_asynccnt)
#define USE_ASYNC_LDS 1
typedef __attribute__((address_space(1))) v4i g_v4i;   // global (AS1)
typedef __attribute__((address_space(3))) v4i l_v4i;   // LDS (AS3)
#else
#define USE_ASYNC_LDS 0
#endif

// ---- fragment helpers ------------------------------------------------------

// Load a 16-f16 A/B fragment for the 16x16x32 WMMA from LDS.
// CDNA5 16-bit A layout: lane L holds M=L&15; lanes<16 own K[0..7]/K[16..23],
// lanes>=16 own K[8..15]/K[24..31] -> two 16B chunks at +0 and +16 elements.
__device__ __forceinline__ v16h load_frag(const _Float16* p) {
  v8h lo = *(const v8h*)p;
  v8h hi = *(const v8h*)(p + 16);
  return __builtin_shufflevector(lo, hi, 0, 1, 2, 3, 4, 5, 6, 7,
                                         8, 9, 10, 11, 12, 13, 14, 15);
}

__device__ __forceinline__ v8f wmma16(v16h a, v16h b, v8f c) {
  return __builtin_amdgcn_wmma_f32_16x16x32_f16(
      /*neg_a=*/false, a, /*neg_b=*/false, b,
      /*c_mod=*/(short)0, c, /*reuse_a=*/false, /*reuse_b=*/false);
}

// ---- activations -----------------------------------------------------------

template <int ACT>
__device__ __forceinline__ float act_apply(float x) {
  if (ACT == 1) {                       // SiLU
    return x / (1.0f + __expf(-x));
  } else if (ACT == 2) {                // Mish = x * tanh(softplus(x))
    float sp = (x > 20.0f) ? x : log1pf(__expf(x));
    return x * tanhf(sp);
  }
  return x;
}

// ---------------------------------------------------------------------------
// Generic WMMA GEMM:  C[M,N] = act(A[M,K] @ W[K,N] + bias[N])
// fp32 in/out; staged to LDS as f16, double buffered. Block tile 128x128,
// BK=32. 256 threads = 8 waves in 2x4; wave tile 64x32 = 4x2 accumulators.
// Requires M % 128 == 0, K % 32 == 0; N % 4 == 0 (N < tile is guarded).
// ---------------------------------------------------------------------------
template <int ACT>
__global__ __launch_bounds__(256, 2) void gemm_f16_wmma(
    const float* __restrict__ A, const float* __restrict__ W,
    const float* __restrict__ bias, float* __restrict__ C,
    int M, int N, int K) {
  __shared__ __align__(16) _Float16 As[2][128 * 40];  // [m][k], k pad 32->40
  __shared__ __align__(16) _Float16 Bs[2][128 * 40];  // [n][k] (transposed)

  const int tid  = threadIdx.x;
  const int m0   = blockIdx.x * 128;
  const int n0   = blockIdx.y * 128;
  const int wave = tid >> 5;
  const int lane = tid & 31;
  const int wr   = (wave & 1) << 6;      // 0 / 64
  const int wc   = (wave >> 1) << 5;     // 0 / 32 / 64 / 96
  const int rr   = lane & 15;
  const int hi8  = (lane >> 4) << 3;     // 0 or 8
  const int steps = K >> 5;

  v8f acc[4][2];
#pragma unroll
  for (int i = 0; i < 4; ++i)
#pragma unroll
    for (int j = 0; j < 2; ++j) acc[i][j] = (v8f){};

  auto stageA = [&](int buf, int kt) {
#pragma unroll
    for (int i = 0; i < 4; ++i) {
      int idx = tid + (i << 8);          // 1024 float4 = 128x32 tile
      int r   = idx >> 3;
      int c   = (idx & 7) << 2;
      const float4 f =
          *(const float4*)(A + (size_t)(m0 + r) * K + (kt << 5) + c);
      v4h h = {(_Float16)f.x, (_Float16)f.y, (_Float16)f.z, (_Float16)f.w};
      *(v4h*)&As[buf][r * 40 + c] = h;
    }
  };
  auto stageB = [&](int buf, int kt) {
#pragma unroll
    for (int i = 0; i < 4; ++i) {
      int idx = tid + (i << 8);          // 1024 float4 = 32x128 tile
      int kr  = idx >> 5;                // K row 0..31
      int c   = (idx & 31) << 2;         // N col 0..124
      float4 f = make_float4(0.f, 0.f, 0.f, 0.f);
      if (n0 + c < N)
        f = *(const float4*)(W + (size_t)((kt << 5) + kr) * N + n0 + c);
      _Float16* d = &Bs[buf][c * 40 + kr];
      d[0]   = (_Float16)f.x;
      d[40]  = (_Float16)f.y;
      d[80]  = (_Float16)f.z;
      d[120] = (_Float16)f.w;
    }
  };

  stageA(0, 0);
  stageB(0, 0);
  __syncthreads();

  for (int kt = 0; kt < steps; ++kt) {
    const int buf = kt & 1;
    if (kt + 2 < steps) {  // global_prefetch_b8 on gfx1250
      __builtin_prefetch(A + (size_t)(m0 + (tid >> 3)) * K + ((kt + 2) << 5),
                         0, 1);
      __builtin_prefetch(W + (size_t)(((kt + 2) << 5) + (tid >> 5)) * N + n0,
                         0, 1);
    }
    if (kt + 1 < steps) {
      stageA(buf ^ 1, kt + 1);
      stageB(buf ^ 1, kt + 1);
    }

    v16h a[4], b[2];
#pragma unroll
    for (int i = 0; i < 4; ++i)
      a[i] = load_frag(&As[buf][(wr + i * 16 + rr) * 40 + hi8]);
#pragma unroll
    for (int j = 0; j < 2; ++j)
      b[j] = load_frag(&Bs[buf][(wc + j * 16 + rr) * 40 + hi8]);
#pragma unroll
    for (int i = 0; i < 4; ++i)
#pragma unroll
      for (int j = 0; j < 2; ++j) acc[i][j] = wmma16(a[i], b[j], acc[i][j]);
    __syncthreads();
  }

  // Epilogue. C/D layout: lane holds N = lane&15; VGPR r holds M = r + hi8.
#pragma unroll
  for (int j = 0; j < 2; ++j) {
    const int cn = n0 + wc + j * 16 + rr;
    if (cn >= N) continue;
    const float bv = bias ? bias[cn] : 0.f;
#pragma unroll
    for (int i = 0; i < 4; ++i) {
      const int rbase = m0 + wr + i * 16 + hi8;
#pragma unroll
      for (int r = 0; r < 8; ++r)
        C[(size_t)(rbase + r) * N + cn] = act_apply<ACT>(acc[i][j][r] + bv);
    }
  }
}

// ---------------------------------------------------------------------------
// Fused attention per (b,h): S = QK^T/sqrt(32) + smolgen_bias; P = softmax(S);
// AV = P @ V. 256 thr / 8 waves per block, one block per head instance.
// The 16KB smolgen bias tile is pulled in with async global->LDS copies
// (ASYNCcnt) overlapping the Q/K/V staging when the builtin is available.
// ---------------------------------------------------------------------------
__global__ __launch_bounds__(256, 2) void attn_fused(
    const float* __restrict__ Q, const float* __restrict__ Kmat,
    const float* __restrict__ V, const float* __restrict__ SW,
    float* __restrict__ AV) {
  __shared__ __align__(16) _Float16 Qs[64 * 40];  // [q][d]
  __shared__ __align__(16) _Float16 Ks[64 * 40];  // [k][d]
  __shared__ __align__(16) _Float16 Vt[32 * 72];  // [d][k] (transposed)
  __shared__ __align__(16) float    SWs[SEQ_LEN * SEQ_LEN];  // smolgen bias
  __shared__ __align__(16) float    Sf[64 * 68];  // logits f32
  __shared__ __align__(16) _Float16 Ps[64 * 72];  // probs  f16

  const int bh  = blockIdx.x;
  const int b   = bh >> 5;
  const int h   = bh & (H_HEADS - 1);
  const int tid = threadIdx.x;
  const size_t base = (size_t)b * SEQ_LEN * D_DIM + (size_t)h * HD_DIM;
  const float* swg = SW + (size_t)bh * (SEQ_LEN * SEQ_LEN);

  // Kick off the smolgen-bias tile copy first so it overlaps QKV staging.
#if USE_ASYNC_LDS
#pragma unroll
  for (int i = 0; i < 4; ++i) {
    const int idx = (tid + (i << 8)) << 2;   // 4 floats per b128
    __builtin_amdgcn_global_load_async_to_lds_b128(
        (g_v4i*)(swg + idx), (l_v4i*)(&SWs[idx]), 0, 0);
  }
#else
#pragma unroll
  for (int i = 0; i < 4; ++i) {
    const int idx = (tid + (i << 8)) << 2;
    *(float4*)&SWs[idx] = *(const float4*)(swg + idx);
  }
#endif

#pragma unroll
  for (int i = 0; i < 2; ++i) {
    int idx = tid + (i << 8);            // 512 float4 per tensor
    int s   = idx >> 3;                  // seq position 0..63
    int c   = (idx & 7) << 2;            // head-dim 0..28
    float4 fq = *(const float4*)(Q + base + (size_t)s * D_DIM + c);
    float4 fk = *(const float4*)(Kmat + base + (size_t)s * D_DIM + c);
    float4 fv = *(const float4*)(V + base + (size_t)s * D_DIM + c);
    *(v4h*)&Qs[s * 40 + c] =
        (v4h){(_Float16)fq.x, (_Float16)fq.y, (_Float16)fq.z, (_Float16)fq.w};
    *(v4h*)&Ks[s * 40 + c] =
        (v4h){(_Float16)fk.x, (_Float16)fk.y, (_Float16)fk.z, (_Float16)fk.w};
    Vt[(c + 0) * 72 + s] = (_Float16)fv.x;
    Vt[(c + 1) * 72 + s] = (_Float16)fv.y;
    Vt[(c + 2) * 72 + s] = (_Float16)fv.z;
    Vt[(c + 3) * 72 + s] = (_Float16)fv.w;
  }
#if USE_ASYNC_LDS
  __builtin_amdgcn_s_wait_asynccnt(0);  // own copies done; barrier publishes
#endif
  __syncthreads();

  const int wave = tid >> 5, lane = tid & 31;
  const int rr = lane & 15, hi8 = (lane >> 4) << 3;

  {  // S = Q @ K^T  (M=64 q, N=64 k, contraction d=32 -> single WMMA step)
    const int mT = wave & 3, nB = (wave >> 2) << 1;
    v16h a  = load_frag(&Qs[(mT * 16 + rr) * 40 + hi8]);
    v16h b0 = load_frag(&Ks[(nB * 16 + rr) * 40 + hi8]);
    v16h b1 = load_frag(&Ks[((nB + 1) * 16 + rr) * 40 + hi8]);
    v8f s0 = {}, s1 = {};
    s0 = wmma16(a, b0, s0);
    s1 = wmma16(a, b1, s1);
    const int cn0 = nB * 16 + rr;
    const int cn1 = cn0 + 16;
    const int rb  = mT * 16 + hi8;
#pragma unroll
    for (int r = 0; r < 8; ++r) {
      Sf[(rb + r) * 68 + cn0] =
          s0[r] * QK_SCALE + SWs[(rb + r) * SEQ_LEN + cn0];
      Sf[(rb + r) * 68 + cn1] =
          s1[r] * QK_SCALE + SWs[(rb + r) * SEQ_LEN + cn1];
    }
  }
  __syncthreads();

  if (tid < SEQ_LEN) {  // row softmax
    const int row = tid;
    float m = -3.4e38f;
    for (int j = 0; j < SEQ_LEN; ++j) m = fmaxf(m, Sf[row * 68 + j]);
    float sum = 0.f;
    for (int j = 0; j < SEQ_LEN; ++j) {
      float e = __expf(Sf[row * 68 + j] - m);
      Sf[row * 68 + j] = e;
      sum += e;
    }
    float inv = 1.f / sum;
    for (int j = 0; j < SEQ_LEN; ++j)
      Ps[row * 72 + j] = (_Float16)(Sf[row * 68 + j] * inv);
  }
  __syncthreads();

  {  // AV = P @ V  (M=64 q, N=32 d, contraction k=64 -> 2 WMMA steps)
    const int mT = wave & 3, nT = wave >> 2;  // 8 waves -> 4x2 tiles
    v8f acc = {};
#pragma unroll
    for (int st = 0; st < 2; ++st) {
      const int kb = st << 5;
      v16h a  = load_frag(&Ps[(mT * 16 + rr) * 72 + kb + hi8]);
      v16h bb = load_frag(&Vt[(nT * 16 + rr) * 72 + kb + hi8]);
      acc = wmma16(a, bb, acc);
    }
    const int cn = nT * 16 + rr;
    const int rb = mT * 16 + hi8;
#pragma unroll
    for (int r = 0; r < 8; ++r)
      AV[base + (size_t)(rb + r) * D_DIM + cn] = acc[r];
  }
}

// ---------------------------------------------------------------------------
// LayerNorm over rows: out = (in - mean) * rsqrt(var + 1e-3) * g + b
// ---------------------------------------------------------------------------
__global__ __launch_bounds__(256) void ln_rows(
    const float* __restrict__ in, const float* __restrict__ g,
    const float* __restrict__ b, float* __restrict__ out, int width) {
  __shared__ float s1[256], s2[256];
  const int row = blockIdx.x, tid = threadIdx.x;
  const float* r = in + (size_t)row * width;
  float a = 0.f, aa = 0.f;
  for (int i = tid; i < width; i += 256) {
    float v = r[i];
    a += v;
    aa += v * v;
  }
  s1[tid] = a;
  s2[tid] = aa;
  __syncthreads();
  for (int o = 128; o > 0; o >>= 1) {
    if (tid < o) {
      s1[tid] += s1[tid + o];
      s2[tid] += s2[tid + o];
    }
    __syncthreads();
  }
  const float mean = s1[0] / (float)width;
  const float var  = s2[0] / (float)width - mean * mean;
  const float inv  = rsqrtf(var + 1e-3f);
  float* o = out + (size_t)row * width;
  for (int i = tid; i < width; i += 256)
    o[i] = (r[i] - mean) * inv * g[i] + b[i];
}

// out = LN(x + alpha*y) with gain/bias
__global__ __launch_bounds__(256) void res_ln_rows(
    const float* __restrict__ x, const float* __restrict__ y,
    const float* __restrict__ g, const float* __restrict__ b,
    float* __restrict__ out, int width, float alpha) {
  __shared__ float s1[256], s2[256];
  const int row = blockIdx.x, tid = threadIdx.x;
  const float* xr = x + (size_t)row * width;
  const float* yr = y + (size_t)row * width;
  float a = 0.f, aa = 0.f;
  for (int i = tid; i < width; i += 256) {
    float v = xr[i] + alpha * yr[i];
    a += v;
    aa += v * v;
  }
  s1[tid] = a;
  s2[tid] = aa;
  __syncthreads();
  for (int o = 128; o > 0; o >>= 1) {
    if (tid < o) {
      s1[tid] += s1[tid + o];
      s2[tid] += s2[tid + o];
    }
    __syncthreads();
  }
  const float mean = s1[0] / (float)width;
  const float var  = s2[0] / (float)width - mean * mean;
  const float inv  = rsqrtf(var + 1e-3f);
  float* o = out + (size_t)row * width;
  for (int i = tid; i < width; i += 256) {
    float v = xr[i] + alpha * yr[i];
    o[i] = (v - mean) * inv * g[i] + b[i];
  }
}

// ---------------------------------------------------------------------------
// Host-side orchestration
// ---------------------------------------------------------------------------
extern "C" void kernel_launch(void* const* d_in, const int* in_sizes, int n_in,
                              void* d_out, int out_size, void* d_ws,
                              size_t ws_size, hipStream_t stream) {
  (void)in_sizes; (void)n_in; (void)out_size; (void)ws_size;

  const float* X     = (const float*)d_in[0];
  const float* wq_w  = (const float*)d_in[1];
  const float* wq_b  = (const float*)d_in[2];
  const float* wk_w  = (const float*)d_in[3];
  const float* wk_b  = (const float*)d_in[4];
  const float* wv_w  = (const float*)d_in[5];
  const float* wv_b  = (const float*)d_in[6];
  const float* wo_w  = (const float*)d_in[7];
  const float* wo_b  = (const float*)d_in[8];
  const float* sc_w  = (const float*)d_in[9];
  const float* sh1_w = (const float*)d_in[10];
  const float* sh1_b = (const float*)d_in[11];
  const float* sln1g = (const float*)d_in[12];
  const float* sln1b = (const float*)d_in[13];
  const float* sgf_w = (const float*)d_in[14];
  const float* sgf_b = (const float*)d_in[15];
  const float* sln2g = (const float*)d_in[16];
  const float* sln2b = (const float*)d_in[17];
  const float* swg_w = (const float*)d_in[18];
  const float* f1_w  = (const float*)d_in[19];
  const float* f1_b  = (const float*)d_in[20];
  const float* f2_w  = (const float*)d_in[21];
  const float* f2_b  = (const float*)d_in[22];
  const float* n1_g  = (const float*)d_in[23];
  const float* n1_b  = (const float*)d_in[24];
  const float* n2_g  = (const float*)d_in[25];
  const float* n2_b  = (const float*)d_in[26];

  // Workspace carve-up (all f32).
  char*  ws  = (char*)d_ws;
  size_t off = 0;
  auto alloc = [&](size_t nfloats) {
    float* p = (float*)(ws + off);
    off += nfloats * sizeof(float);
    return p;
  };
  float* qb  = alloc((size_t)ROWS * D_DIM);
  float* kb  = alloc((size_t)ROWS * D_DIM);
  float* vb  = alloc((size_t)ROWS * D_DIM);
  float* smc = alloc((size_t)ROWS * SC_DIM);          // == [B, 2048]
  float* smh = alloc((size_t)B_BATCH * SH_DIM);
  float* smh2= alloc((size_t)B_BATCH * SH_DIM);
  float* smg = alloc((size_t)B_BATCH * H_HEADS * SG_DIM);
  float* smg2= alloc((size_t)B_BATCH * H_HEADS * SG_DIM);
  float* swb = alloc((size_t)B_BATCH * H_HEADS * SEQ_LEN * SEQ_LEN);
  float* avb = alloc((size_t)ROWS * D_DIM);
  float* pj  = alloc((size_t)ROWS * E_DIM);
  float* o1  = alloc((size_t)ROWS * E_DIM);
  float* fh  = alloc((size_t)ROWS * DFF_DIM);
  float* fo  = alloc((size_t)ROWS * E_DIM);
  float* xa  = alloc((size_t)ROWS * E_DIM);
  float* xb  = alloc((size_t)ROWS * E_DIM);

  const dim3 blk(256);
  const float* xcur = X;

  for (int l = 0; l < L_LAYERS; ++l) {
    const float* Wq = wq_w + (size_t)l * E_DIM * D_DIM;
    const float* Bq = wq_b + (size_t)l * D_DIM;
    const float* Wk = wk_w + (size_t)l * E_DIM * D_DIM;
    const float* Bk = wk_b + (size_t)l * D_DIM;
    const float* Wv = wv_w + (size_t)l * E_DIM * D_DIM;
    const float* Bv = wv_b + (size_t)l * D_DIM;
    const float* Wo = wo_w + (size_t)l * D_DIM * E_DIM;
    const float* Bo = wo_b + (size_t)l * E_DIM;
    const float* Sc = sc_w + (size_t)l * E_DIM * SC_DIM;
    const float* Sh = sh1_w + (size_t)l * (SEQ_LEN * SC_DIM) * SH_DIM;
    const float* Shb= sh1_b + (size_t)l * SH_DIM;
    const float* L1g= sln1g + (size_t)l * SH_DIM;
    const float* L1b= sln1b + (size_t)l * SH_DIM;
    const float* Sg = sgf_w + (size_t)l * SH_DIM * (H_HEADS * SG_DIM);
    const float* Sgb= sgf_b + (size_t)l * (H_HEADS * SG_DIM);
    const float* L2g= sln2g + (size_t)l * (H_HEADS * SG_DIM);
    const float* L2b= sln2b + (size_t)l * (H_HEADS * SG_DIM);
    const float* Sw = swg_w + (size_t)l * SG_DIM * (SEQ_LEN * SEQ_LEN);
    const float* F1 = f1_w + (size_t)l * E_DIM * DFF_DIM;
    const float* F1b= f1_b + (size_t)l * DFF_DIM;
    const float* F2 = f2_w + (size_t)l * DFF_DIM * E_DIM;
    const float* F2b= f2_b + (size_t)l * E_DIM;
    const float* N1g= n1_g + (size_t)l * E_DIM;
    const float* N1b= n1_b + (size_t)l * E_DIM;
    const float* N2g= n2_g + (size_t)l * E_DIM;
    const float* N2b= n2_b + (size_t)l * E_DIM;

    // QKV projections: [8192,1024] = x @ [1024,1024] + b
    gemm_f16_wmma<0><<<dim3(ROWS / 128, D_DIM / 128), blk, 0, stream>>>(
        xcur, Wq, Bq, qb, ROWS, D_DIM, E_DIM);
    gemm_f16_wmma<0><<<dim3(ROWS / 128, D_DIM / 128), blk, 0, stream>>>(
        xcur, Wk, Bk, kb, ROWS, D_DIM, E_DIM);
    gemm_f16_wmma<0><<<dim3(ROWS / 128, D_DIM / 128), blk, 0, stream>>>(
        xcur, Wv, Bv, vb, ROWS, D_DIM, E_DIM);

    // smolgen: c = x @ sc_w  -> [8192,32] == [B, 2048]
    gemm_f16_wmma<0><<<dim3(ROWS / 128, 1), blk, 0, stream>>>(
        xcur, Sc, nullptr, smc, ROWS, SC_DIM, E_DIM);
    // h = LN(silu(c @ sh1_w + b))   [128, 256]
    gemm_f16_wmma<1><<<dim3(1, SH_DIM / 128), blk, 0, stream>>>(
        smc, Sh, Shb, smh, B_BATCH, SH_DIM, SEQ_LEN * SC_DIM);
    ln_rows<<<B_BATCH, blk, 0, stream>>>(smh, L1g, L1b, smh2, SH_DIM);
    // g = LN(silu(h @ sgf_w + b))   [128, 8192]
    gemm_f16_wmma<1><<<dim3(1, (H_HEADS * SG_DIM) / 128), blk, 0, stream>>>(
        smh2, Sg, Sgb, smg, B_BATCH, H_HEADS * SG_DIM, SH_DIM);
    ln_rows<<<B_BATCH, blk, 0, stream>>>(smg, L2g, L2b, smg2,
                                         H_HEADS * SG_DIM);
    // sw = g.view(4096,256) @ swg_w(256,4096)  -> [B*H, 64*64]
    gemm_f16_wmma<0><<<dim3((B_BATCH * H_HEADS) / 128,
                            (SEQ_LEN * SEQ_LEN) / 128), blk, 0, stream>>>(
        smg2, Sw, nullptr, swb, B_BATCH * H_HEADS, SEQ_LEN * SEQ_LEN, SG_DIM);

    // fused attention per (b,h)
    attn_fused<<<B_BATCH * H_HEADS, blk, 0, stream>>>(qb, kb, vb, swb, avb);

    // output projection + residual LN
    gemm_f16_wmma<0><<<dim3(ROWS / 128, E_DIM / 128), blk, 0, stream>>>(
        avb, Wo, Bo, pj, ROWS, E_DIM, D_DIM);
    res_ln_rows<<<ROWS, blk, 0, stream>>>(xcur, pj, N1g, N1b, o1, E_DIM,
                                          ALPHA_RES);

    // FFN: mish(o1 @ F1 + b1) @ F2 + b2, then residual LN
    gemm_f16_wmma<2><<<dim3(ROWS / 128, DFF_DIM / 128), blk, 0, stream>>>(
        o1, F1, F1b, fh, ROWS, DFF_DIM, E_DIM);
    gemm_f16_wmma<0><<<dim3(ROWS / 128, E_DIM / 128), blk, 0, stream>>>(
        fh, F2, F2b, fo, ROWS, E_DIM, DFF_DIM);

    float* xnext = (l == L_LAYERS - 1) ? (float*)d_out : ((l & 1) ? xb : xa);
    res_ln_rows<<<ROWS, blk, 0, stream>>>(o1, fo, N2g, N2b, xnext, E_DIM,
                                          ALPHA_RES);
    xcur = xnext;
  }
}
